// MetaKAN_50912542327493
// MI455X (gfx1250) — matrix operand order repaired and checked by reading, not compiled, not run on hardware
//
#include <hip/hip_runtime.h>

// ---------------------------------------------------------------------------
// Types for gfx1250 WMMA (wave32): v_wmma_f32_16x16x32_bf16
// ---------------------------------------------------------------------------
typedef __attribute__((ext_vector_type(16))) __bf16 v16bf;
typedef __attribute__((ext_vector_type(8)))  __bf16 v8bf;
typedef __attribute__((ext_vector_type(8)))  float  v8f;

union v16bf_u { v16bf v; v8bf h[2]; };

__device__ __forceinline__ v8f wmma_bf16(v16bf a, v16bf b, v8f c) {
  // (neg_a, A, neg_b, B, c_mod, C, reuse_a, reuse_b)
  return __builtin_amdgcn_wmma_f32_16x16x32_bf16(false, a, false, b, (short)0, c,
                                                 false, false);
}

__device__ __forceinline__ v16bf cvt16(v8f a, v8f b) {
  v16bf r;
#pragma unroll
  for (int i = 0; i < 8; ++i) {
    r[i]     = (__bf16)a[i];
    r[i + 8] = (__bf16)b[i];
  }
  return r;
}

// Problem constants
#define NN   4096
#define INF  512
#define OUTF 512
#define KDIM (INF * 9)          // 4608 = in_f * D_B
#define RROWS (INF * OUTF)      // 262144 emb rows per layer

// ---------------------------------------------------------------------------
// Kernel 1: fused weight generation
//   W(o, i*9+s) = ( (emb @ w1^T + b1) @ w2^T + b2 )[r = o*512+i, s]   (bf16)
// Stage 1: per-wave 16x128 tile = emb(16x64) @ w1^T(64x128)  -> 16 WMMA
// Stage 2: tile(16x128) @ w2^T(128x16, cols 9..15 zero)      ->  4 WMMA
// ---------------------------------------------------------------------------
__global__ __launch_bounds__(256) void wgen_kernel(
    const float* __restrict__ emb, const float* __restrict__ w1,
    const float* __restrict__ b1,  const float* __restrict__ w2,
    const float* __restrict__ b2,  __bf16* __restrict__ Wout, int rows) {
  __shared__ __align__(32) __bf16 lds_t[8][16 * 128];   // 32 KB

  const int lane  = threadIdx.x & 31;
  const int wave  = threadIdx.x >> 5;
  const int rbase = blockIdx.x * 128 + wave * 16;
  if (rbase >= rows) return;

  const int l15   = lane & 15;
  const int ahalf = (lane >> 4) << 3;   // +8 M (accumulator row) for upper half-lanes
  const int bhalf = (lane >> 4) << 4;   // +16 K for upper half-lanes (B operand)

  // ---- Stage 1: t = emb @ w1^T  (M=16 per wave, K=64, N=128) ----
  v8f acc[8];
#pragma unroll
  for (int nt = 0; nt < 8; ++nt)
#pragma unroll
    for (int r = 0; r < 8; ++r) acc[nt][r] = 0.0f;

#pragma unroll
  for (int kb = 0; kb < 2; ++kb) {
    const int k0 = kb * 32;
    // A fragment: row rbase + l15, K = k0+bhalf .. (+0..15 contiguous)
    const float* pe = emb + (size_t)(rbase + l15) * 64 + (k0 + bhalf);
    v8f e0 = *(const v8f*)(pe);
    v8f e1 = *(const v8f*)(pe + 8);
    v16bf afrag = cvt16(e0, e1);
#pragma unroll
    for (int nt = 0; nt < 8; ++nt) {
      // B fragment: column n = nt*16 + l15 of w1^T == row of w1 (contiguous K)
      const float* pw = w1 + (size_t)(nt * 16 + l15) * 64 + (k0 + bhalf);
      v8f f0 = *(const v8f*)(pw);
      v8f f1 = *(const v8f*)(pw + 8);
      acc[nt] = wmma_bf16(afrag, cvt16(f0, f1), acc[nt]);
    }
  }

  // ---- bias b1 + re-layout through LDS (C layout -> A layout) ----
#pragma unroll
  for (int nt = 0; nt < 8; ++nt) {
    const int n = nt * 16 + l15;        // N index (valid for both lane halves)
    const float bias = b1[n];
#pragma unroll
    for (int r = 0; r < 8; ++r) {
      // C layout: vgpr r -> M = r (lanes 0-15) / r+8 (lanes 16-31)
      lds_t[wave][(r + ahalf) * 128 + n] = (__bf16)(acc[nt][r] + bias);
    }
  }
  __syncthreads();

  // ---- Stage 2: out16 = t(16x128) @ w2^T(128x16, padded) ----
  v8f acc2;
#pragma unroll
  for (int r = 0; r < 8; ++r) acc2[r] = 0.0f;

#pragma unroll
  for (int kb = 0; kb < 4; ++kb) {
    const int k0 = kb * 32;
    v16bf_u afrag;
    const __bf16* pt = &lds_t[wave][l15 * 128 + (k0 + bhalf)];
    afrag.h[0] = *(const v8bf*)(pt);
    afrag.h[1] = *(const v8bf*)(pt + 8);

    v16bf bfrag;
    if (l15 < 9) {
      const float* pw = w2 + (size_t)l15 * 128 + (k0 + bhalf);
      v8f f0 = *(const v8f*)(pw);
      v8f f1 = *(const v8f*)(pw + 8);
      bfrag = cvt16(f0, f1);
    } else {
#pragma unroll
      for (int i = 0; i < 16; ++i) bfrag[i] = (__bf16)0.0f;
    }
    acc2 = wmma_bf16(afrag.v, bfrag, acc2);
  }

  // ---- bias b2 + store bf16 W rows (flat index r*9 + s == W(o, i*9+s)) ----
  if (l15 < 9) {
    const float bias2 = b2[l15];
#pragma unroll
    for (int r = 0; r < 8; ++r) {
      const int rr = rbase + r + ahalf;
      Wout[(size_t)rr * 9 + l15] = (__bf16)(acc2[r] + bias2);
    }
  }
}

// ---------------------------------------------------------------------------
// Kernel 2: activation matrix  A[n, i*9+s] = spline basis (s<8) / silu (s==8)
// Literal Cox–de Boor recursion matching the reference (incl. eps=1e-8).
// ---------------------------------------------------------------------------
__global__ __launch_bounds__(256) void act_kernel(const float* __restrict__ h,
                                                  __bf16* __restrict__ A,
                                                  int total) {
  const int idx = blockIdx.x * blockDim.x + threadIdx.x;
  if (idx >= total) return;
  const float x = h[idx];

  float g[12];
#pragma unroll
  for (int t = 0; t < 12; ++t) g[t] = (float)(t - 3) * 0.4f - 1.0f;

  float b[11];
#pragma unroll
  for (int t = 0; t < 11; ++t)
    b[t] = (x >= g[t] && x < g[t + 1]) ? 1.0f : 0.0f;

  const float eps = 1e-8f;
#pragma unroll
  for (int k = 1; k <= 3; ++k) {
#pragma unroll
    for (int t = 0; t < 11; ++t) {   // in-place forward update, valid t < 11-k
      if (t < 11 - k) {
        const float dp = g[t + k] - g[t];
        const float dn = g[t + k + 1] - g[t + 1];
        b[t] = (x - g[t]) / (dp + eps) * b[t] +
               (g[t + k + 1] - x) / (dn + eps) * b[t + 1];
      }
    }
  }

  const float silu = x / (1.0f + __expf(-x));

  const size_t off = (size_t)idx * 9;
#pragma unroll
  for (int s = 0; s < 8; ++s) A[off + s] = (__bf16)b[s];
  A[off + 8] = (__bf16)silu;
}

// ---------------------------------------------------------------------------
// Kernel 3: main GEMM  C(M x N) = A(M x K) @ W(N x K)^T, bf16 in, fp32 out.
// 4 waves/block; each wave owns a 32(M) x 64(N) tile = 2x4 WMMA accumulators.
// ---------------------------------------------------------------------------
__global__ __launch_bounds__(128) void gemm_kernel(
    const __bf16* __restrict__ A, const __bf16* __restrict__ W,
    float* __restrict__ C, int M, int Nn, int K) {
  const int lane = threadIdx.x & 31;
  const int wave = threadIdx.x >> 5;
  const int m0 = blockIdx.x * 128 + wave * 32;
  const int n0 = blockIdx.y * 64;

  const int l15   = lane & 15;
  const int ahalf = (lane >> 4) << 3;   // C layout: +8 M on upper half-lanes
  const int bhalf = (lane >> 4) << 4;   // B operand: +16 K on upper half-lanes

  v8f acc[2][4];
#pragma unroll
  for (int i = 0; i < 2; ++i)
#pragma unroll
    for (int j = 0; j < 4; ++j)
#pragma unroll
      for (int r = 0; r < 8; ++r) acc[i][j][r] = 0.0f;

  const __bf16* pa0 = A + (size_t)(m0 + l15) * K;
  const __bf16* pa1 = A + (size_t)(m0 + 16 + l15) * K;
  const __bf16* pb0 = W + (size_t)(n0 + l15) * K;        // B column = W row
  const __bf16* pb1 = W + (size_t)(n0 + 16 + l15) * K;
  const __bf16* pb2 = W + (size_t)(n0 + 32 + l15) * K;
  const __bf16* pb3 = W + (size_t)(n0 + 48 + l15) * K;

  for (int k0 = 0; k0 < K; k0 += 32) {
    const int ka = k0 + bhalf;
    const int kb = k0 + bhalf;

    __builtin_prefetch(pa0 + ka + 128, 0, 1);
    __builtin_prefetch(pb0 + kb + 128, 0, 1);

    v16bf_u a0, a1;
    a0.h[0] = *(const v8bf*)(pa0 + ka);
    a0.h[1] = *(const v8bf*)(pa0 + ka + 8);
    a1.h[0] = *(const v8bf*)(pa1 + ka);
    a1.h[1] = *(const v8bf*)(pa1 + ka + 8);

    v16bf b0 = *(const v16bf*)(pb0 + kb);
    v16bf b1 = *(const v16bf*)(pb1 + kb);
    v16bf b2 = *(const v16bf*)(pb2 + kb);
    v16bf b3 = *(const v16bf*)(pb3 + kb);

    acc[0][0] = wmma_bf16(a0.v, b0, acc[0][0]);
    acc[1][0] = wmma_bf16(a1.v, b0, acc[1][0]);
    acc[0][1] = wmma_bf16(a0.v, b1, acc[0][1]);
    acc[1][1] = wmma_bf16(a1.v, b1, acc[1][1]);
    acc[0][2] = wmma_bf16(a0.v, b2, acc[0][2]);
    acc[1][2] = wmma_bf16(a1.v, b2, acc[1][2]);
    acc[0][3] = wmma_bf16(a0.v, b3, acc[0][3]);
    acc[1][3] = wmma_bf16(a1.v, b3, acc[1][3]);
  }

#pragma unroll
  for (int i = 0; i < 2; ++i)
#pragma unroll
    for (int j = 0; j < 4; ++j)
#pragma unroll
      for (int r = 0; r < 8; ++r) {
        const int row = m0 + i * 16 + r + ahalf;   // C layout M index
        const int col = n0 + j * 16 + l15;         // C layout N index
        C[(size_t)row * Nn + col] = acc[i][j][r];
      }
}

// ---------------------------------------------------------------------------
// Host launcher
// ---------------------------------------------------------------------------
extern "C" void kernel_launch(void* const* d_in, const int* in_sizes, int n_in,
                              void* d_out, int out_size, void* d_ws,
                              size_t ws_size, hipStream_t stream) {
  const float* x    = (const float*)d_in[0];
  const float* emb0 = (const float*)d_in[1];
  const float* w1_0 = (const float*)d_in[2];
  const float* b1_0 = (const float*)d_in[3];
  const float* w2_0 = (const float*)d_in[4];
  const float* b2_0 = (const float*)d_in[5];
  const float* emb1 = (const float*)d_in[6];
  const float* w1_1 = (const float*)d_in[7];
  const float* b1_1 = (const float*)d_in[8];
  const float* w2_1 = (const float*)d_in[9];
  const float* b2_1 = (const float*)d_in[10];

  // Workspace: A (bf16, N x K) | W (bf16, OUT x K) | H (fp32, N x HID)
  const size_t A_bytes = (size_t)NN * KDIM * 2;      // 37,748,736
  const size_t W_bytes = (size_t)OUTF * KDIM * 2;    //  4,718,592
  char* ws = (char*)d_ws;
  __bf16* Abuf = (__bf16*)(ws);
  __bf16* Wbuf = (__bf16*)(ws + A_bytes);
  float*  Hbuf = (float*)(ws + A_bytes + W_bytes);

  const dim3 gemm_grid(NN / 128, OUTF / 64);   // 32 x 8
  const int  act_blocks = (NN * INF) / 256;    // 8192
  const int  wgen_blocks = RROWS / 128;        // 2048

  // ---- Layer 0 ----
  wgen_kernel<<<wgen_blocks, 256, 0, stream>>>(emb0, w1_0, b1_0, w2_0, b2_0,
                                               Wbuf, RROWS);
  act_kernel<<<act_blocks, 256, 0, stream>>>(x, Abuf, NN * INF);
  gemm_kernel<<<gemm_grid, 128, 0, stream>>>(Abuf, Wbuf, Hbuf, NN, OUTF, KDIM);

  // ---- Layer 1 ----
  wgen_kernel<<<wgen_blocks, 256, 0, stream>>>(emb1, w1_1, b1_1, w2_1, b2_1,
                                               Wbuf, RROWS);
  act_kernel<<<act_blocks, 256, 0, stream>>>(Hbuf, Abuf, NN * INF);
  gemm_kernel<<<gemm_grid, 128, 0, stream>>>(Abuf, Wbuf, (float*)d_out, NN,
                                             OUTF, KDIM);
}
